// LVSSM_27565100105947
// MI455X (gfx1250) — matrix operand order repaired and checked by reading, not compiled, not run on hardware
//
#include <hip/hip_runtime.h>
#include <math.h>

// ---------------------------------------------------------------------------
// VMamba-style light-guided 2D selective scan for gfx1250 (MI455X).
//   * GEMMs (in_proj / x_dbl / out_proj) on v_wmma_f32_16x16x32_f16,
//     4 N-tiles per wave (A-fragment reuse), f16 pixel-major operands so
//     every fragment load is an aligned b128.
//   * Scan is state-parallel on wave32 (16 lanes = 16 states, 2 scans/wave)
//     with shfl_xor butterflies and fully coalesced (bk,l,n) aux streams.
// ---------------------------------------------------------------------------

typedef __attribute__((ext_vector_type(16))) _Float16 v16h;
typedef __attribute__((ext_vector_type(8)))  _Float16 v8h;
typedef __attribute__((ext_vector_type(8)))  float    v8f;

#define D_MODEL 96
#define D_STATE 16
#define D_INNER 192
#define DT_RANK 6
#define NB      2
#define LL      4096      // 64*64
#define PIX     (NB*LL)   // 8192

__device__ __forceinline__ float siluf(float v)     { return v / (1.f + __expf(-v)); }
__device__ __forceinline__ float softplusf(float v) { return v > 20.f ? v : log1pf(__expf(v)); }

// position map for scan direction k: xs[k][l] = xc[dir_pos(k,l)], and the
// scan output at step l for direction k lands at the same position.
__device__ __forceinline__ int dir_pos(int k, int l) {
  switch (k) {
    case 0:  return l;
    case 1:  return ((l & 63) << 6) | (l >> 6);
    case 2:  return (LL - 1) - l;
    default: { int l2 = (LL - 1) - l; return ((l2 & 63) << 6) | (l2 >> 6); }
  }
}

__device__ __forceinline__ v16h joinA(v8h lo, v8h hi) {
  return __builtin_shufflevector(lo, hi, 0,1,2,3,4,5,6,7,8,9,10,11,12,13,14,15);
}

// ---------------------------------------------------------------------------
// Setup kernels: zero accumulator, weight f32->f16, x_proj pad, input transpose
// ---------------------------------------------------------------------------
__global__ __launch_bounds__(256)
void k_zero(float* __restrict__ p, int n) {
  int i = blockIdx.x * blockDim.x + threadIdx.x;
  if (i < n) p[i] = 0.f;
}

__global__ __launch_bounds__(256)
void k_f32tof16(const float* __restrict__ s, _Float16* __restrict__ d, int n) {
  int i = blockIdx.x * blockDim.x + threadIdx.x;
  if (i < n) d[i] = (_Float16)s[i];
}

__global__ __launch_bounds__(256)
void k_xpw_pad(const float* __restrict__ xpw, _Float16* __restrict__ d) {
  int i = blockIdx.x * blockDim.x + threadIdx.x;      // (4,48,192)
  if (i >= 4 * 48 * D_INNER) return;
  int dd = i % D_INNER, r = (i / D_INNER) % 48, k = i / (48 * D_INNER);
  d[i] = (r < DT_RANK + 2 * D_STATE) ? (_Float16)xpw[((size_t)k * 38 + r) * D_INNER + dd]
                                     : (_Float16)0.f;
}

// x (b,96,l) f32 -> xT (pix,96) f16   (pix = b*LL + l)
__global__ __launch_bounds__(256)
void k_xpose_in(const float* __restrict__ x, _Float16* __restrict__ xT) {
  int i = blockIdx.x * blockDim.x + threadIdx.x;      // over NB*96*LL
  if (i >= NB * D_MODEL * LL) return;
  int l = i & (LL - 1), c = (i >> 12) % D_MODEL, b = i / (D_MODEL * LL);
  xT[((size_t)b * LL + l) * D_MODEL + c] = (_Float16)x[i];
}

// ---------------------------------------------------------------------------
// Kernel 1: in_proj — xzz(576,pix) = W(576,96) @ x(96,pix) via WMMA.
// Wave = one M-tile x 4 N-tiles.  xx -> channel-major f32 (conv input);
// z1/z2 -> pixel-major f32 (gate input).
// ---------------------------------------------------------------------------
__global__ __launch_bounds__(256)
void k_inproj(const _Float16* __restrict__ xT, const _Float16* __restrict__ w16,
              float* __restrict__ xx, float* __restrict__ z1T, float* __restrict__ z2T)
{
  const int lane = threadIdx.x & 31;
  const int half = lane >> 4, lm = lane & 15;
  const int NG = PIX / 64;                               // 128 N-groups
  int wv = blockIdx.x * 8 + (threadIdx.x >> 5);          // 4608 waves
  const int ng = wv % NG, mtile = wv / NG;               // mtile 0..35
  const int row0 = mtile << 4;
  const int p0 = ng << 6;

  v8f acc[4] = {};
  const _Float16* wrow = w16 + (size_t)(row0 + lm) * D_MODEL;

  for (int kk = 0; kk < D_MODEL; kk += 32) {
    v8h alo = *(const v8h*)(wrow + kk + half * 8);
    v8h ahi = *(const v8h*)(wrow + kk + 16 + half * 8);
    v16h a = joinA(alo, ahi);
#pragma unroll
    for (int t = 0; t < 4; ++t) {
      const _Float16* bp = xT + (size_t)(p0 + t * 16 + lm) * D_MODEL + kk + half * 16;
      __builtin_prefetch(bp + 32, 0, 0);
      v16h bb = joinA(*(const v8h*)(bp - half * 8), *(const v8h*)(bp - half * 8 + 16));
      acc[t] = __builtin_amdgcn_wmma_f32_16x16x32_f16(false, a, false, bb,
                                                      (short)0, acc[t], false, false);
    }
  }
#pragma unroll
  for (int t = 0; t < 4; ++t) {
    int p = p0 + t * 16 + lm;
    int b = p >> 12, l = p & (LL - 1);
#pragma unroll
    for (int j = 0; j < 8; ++j) {
      int o = row0 + j + half * 8;
      float v = acc[t][j];
      if (o < D_INNER)          xx[((size_t)b * D_INNER + o) * LL + l] = v;
      else if (o < 2 * D_INNER) z1T[(size_t)p * D_INNER + (o - D_INNER)]     = v;
      else                      z2T[(size_t)p * D_INNER + (o - 2 * D_INNER)] = v;
    }
  }
}

// ---------------------------------------------------------------------------
// Kernel 2: depthwise 3x3 + bias + SiLU + light prior.
// Emits xc channel-major f32 (scan gather) and xc16 pixel-major f16 (x_dbl).
// ---------------------------------------------------------------------------
__global__ __launch_bounds__(256)
void k_conv(const float* __restrict__ xx, const float* __restrict__ cw,
            const float* __restrict__ cb, const float* __restrict__ lprior,
            const float* __restrict__ lw, const float* __restrict__ lb,
            float* __restrict__ xc, _Float16* __restrict__ xc16)
{
  int idx = blockIdx.x * blockDim.x + threadIdx.x;   // ((b*192+d)*4096+l)
  if (idx >= NB * D_INNER * LL) return;
  int l = idx & (LL - 1);
  int c = idx >> 12;
  int d = c % D_INNER, b = c / D_INNER;
  int hh = l >> 6, ww = l & 63;
  const float* src = xx + (size_t)c * LL;
  const float* wk  = cw + d * 9;
  float s = cb[d];
#pragma unroll
  for (int i = -1; i <= 1; ++i)
#pragma unroll
    for (int j = -1; j <= 1; ++j) {
      int y = hh + i, xq = ww + j;
      if (y >= 0 && y < 64 && xq >= 0 && xq < 64)
        s += src[(y << 6) | xq] * wk[(i + 1) * 3 + (j + 1)];
    }
  float lp = lprior[(size_t)b * LL + l] * lw[d] + lb[d];
  float v = siluf(s) + lp;
  xc[idx] = v;
  xc16[((size_t)b * LL + l) * D_INNER + d] = (_Float16)v;
}

// ---------------------------------------------------------------------------
// Kernel 3: x_dbl — (48,L) = x_proj_pad[k](48,192) @ xs[b,k](192,L) via WMMA.
// Directional gather of xc16 rows; outputs transposed (bk,l,*) for the scan.
// ---------------------------------------------------------------------------
__global__ __launch_bounds__(256)
void k_xdbl(const _Float16* __restrict__ xc16, const _Float16* __restrict__ xpw16,
            float* __restrict__ dtrT, float* __restrict__ BsT, float* __restrict__ CsT)
{
  const int lane = threadIdx.x & 31;
  const int half = lane >> 4, lm = lane & 15;
  int wv = blockIdx.x * 8 + (threadIdx.x >> 5);     // 1536 waves
  const int bk    = wv / 192;                       // b*4+k
  int rem         = wv % 192;
  const int mtile = rem / 64;                       // rows 0..47
  const int ng    = rem % 64;
  const int b = bk >> 2, k = bk & 3;
  const int l0 = ng << 6;
  const int row0 = mtile << 4;

  int pos[4];
#pragma unroll
  for (int t = 0; t < 4; ++t) pos[t] = dir_pos(k, l0 + t * 16 + lm);

  v8f acc[4] = {};
  const _Float16* wrow = xpw16 + ((size_t)k * 48 + row0 + lm) * D_INNER;

  for (int kk = 0; kk < D_INNER; kk += 32) {
    v8h alo = *(const v8h*)(wrow + kk + half * 8);
    v8h ahi = *(const v8h*)(wrow + kk + 16 + half * 8);
    v16h a = joinA(alo, ahi);
#pragma unroll
    for (int t = 0; t < 4; ++t) {
      const _Float16* bp = xc16 + ((size_t)b * LL + pos[t]) * D_INNER + kk + half * 16;
      __builtin_prefetch(bp + 32, 0, 0);
      v16h bb = joinA(*(const v8h*)(bp - half * 8), *(const v8h*)(bp - half * 8 + 16));
      acc[t] = __builtin_amdgcn_wmma_f32_16x16x32_f16(false, a, false, bb,
                                                      (short)0, acc[t], false, false);
    }
  }
#pragma unroll
  for (int t = 0; t < 4; ++t) {
    int l = l0 + t * 16 + lm;
#pragma unroll
    for (int j = 0; j < 8; ++j) {
      int c = row0 + j + half * 8;
      float v = acc[t][j];
      if (c < DT_RANK)
        dtrT[((size_t)bk * LL + l) * DT_RANK + c] = v;
      else if (c < DT_RANK + D_STATE)
        BsT[((size_t)bk * LL + l) * D_STATE + (c - DT_RANK)] = v;
      else if (c < DT_RANK + 2 * D_STATE)
        CsT[((size_t)bk * LL + l) * D_STATE + (c - DT_RANK - D_STATE)] = v;
    }
  }
}

// ---------------------------------------------------------------------------
// Kernel 4: selective scan. 1536 scans (b,k,d); 16 lanes = 16 states, two
// scans per wave. Aux streams are (bk,l,n) so each step is one coalesced
// 64B line per half-wave.  Direction merge fused via atomicAdd at dir_pos
// into the pixel-major accumulator.
// ---------------------------------------------------------------------------
__global__ __launch_bounds__(256)
void k_scan(const float* __restrict__ xc,  const float* __restrict__ dtrT,
            const float* __restrict__ BsT, const float* __restrict__ CsT,
            const float* __restrict__ dtw, const float* __restrict__ dtb,
            const float* __restrict__ Alogs, const float* __restrict__ Dsp,
            float* __restrict__ yaccT)
{
  int tid  = blockIdx.x * blockDim.x + threadIdx.x;
  int wave = tid >> 5, lane = tid & 31;
  int half = lane >> 4, n = lane & 15;
  int s = wave * 2 + half;
  if (s >= NB * 4 * D_INNER) return;
  int b  = s / (4 * D_INNER);
  int k  = (s / D_INNER) & 3;
  int d  = s % D_INNER;
  int bk = b * 4 + k;
  int kd = k * D_INNER + d;

  float An   = -__expf(Alogs[(size_t)kd * D_STATE + n]);
  float bias = dtb[kd];
  float w0 = dtw[kd * DT_RANK + 0], w1 = dtw[kd * DT_RANK + 1];
  float w2 = dtw[kd * DT_RANK + 2], w3 = dtw[kd * DT_RANK + 3];
  float w4 = dtw[kd * DT_RANK + 4], w5 = dtw[kd * DT_RANK + 5];
  float Dv = Dsp[kd];

  const float* xrow = xc   + ((size_t)b * D_INNER + d) * LL;
  const float* dtp  = dtrT + (size_t)bk * LL * DT_RANK;
  const float* Bp   = BsT  + (size_t)bk * LL * D_STATE + n;
  const float* Cp   = CsT  + (size_t)bk * LL * D_STATE + n;
  float*       yb   = yaccT + (size_t)b * LL * D_INNER + d;

  float h = 0.f;
  for (int l = 0; l < LL; ++l) {
    const float* dl = dtp + l * DT_RANK;
    float raw = bias + w0 * dl[0] + w1 * dl[1] + w2 * dl[2]
                     + w3 * dl[3] + w4 * dl[4] + w5 * dl[5];
    float delta = softplusf(raw);
    int pos = dir_pos(k, l);
    float xv = xrow[pos];
    h = __expf(delta * An) * h + delta * xv * Bp[(size_t)l * D_STATE];
    float y = h * Cp[(size_t)l * D_STATE];
    y += __shfl_xor(y, 1);
    y += __shfl_xor(y, 2);
    y += __shfl_xor(y, 4);
    y += __shfl_xor(y, 8);
    if (n == 0) atomicAdd(yb + (size_t)pos * D_INNER, y + Dv * xv);
  }
}

// ---------------------------------------------------------------------------
// Kernel 5: LayerNorm over channels + dual SiLU gates + alpha mix -> g16.
// One wave per pixel; all pixel-major reads are contiguous 192-float rows.
// ---------------------------------------------------------------------------
__global__ __launch_bounds__(256)
void k_lngate(const float* __restrict__ yaccT, const float* __restrict__ z1T,
              const float* __restrict__ z2T,   const float* __restrict__ omap,
              const float* __restrict__ umap,  const float* __restrict__ lw,
              const float* __restrict__ lb,    const float* __restrict__ lnw,
              const float* __restrict__ lnb,   const float* __restrict__ alpha,
              _Float16* __restrict__ g16)
{
  int tid  = blockIdx.x * blockDim.x + threadIdx.x;
  int pix  = tid >> 5, lane = tid & 31;
  if (pix >= PIX) return;
  int b = pix >> 12, l = pix & (LL - 1);
  const size_t base = (size_t)pix * D_INNER;

  float sum = 0.f, sq = 0.f;
#pragma unroll
  for (int j = 0; j < 6; ++j) {
    float v = yaccT[base + lane + 32 * j];
    sum += v; sq += v * v;
  }
#pragma unroll
  for (int m = 16; m >= 1; m >>= 1) {
    sum += __shfl_xor(sum, m);
    sq  += __shfl_xor(sq, m);
  }
  float mu  = sum * (1.f / 192.f);
  float var = sq * (1.f / 192.f) - mu * mu;
  float inv = rsqrtf(var + 1e-5f);
  float al  = alpha[0];
  float ov  = omap[(size_t)b * LL + l];
  float un  = umap[(size_t)b * LL + l];
#pragma unroll
  for (int j = 0; j < 6; ++j) {
    int d = lane + 32 * j;
    float yn = (yaccT[base + d] - mu) * inv * lnw[d] + lnb[d];
    float g1 = yn * siluf(z1T[base + d] + ov * lw[d] + lb[d]);
    float g2 = yn * siluf(z2T[base + d] + un * lw[d] + lb[d]);
    g16[base + d] = (_Float16)(al * g1 + (1.f - al) * g2);
  }
}

// ---------------------------------------------------------------------------
// Kernel 6: out_proj — out(96,pix) = W(96,192) @ g(192,pix) via WMMA.
// ---------------------------------------------------------------------------
__global__ __launch_bounds__(256)
void k_outproj(const _Float16* __restrict__ g16, const _Float16* __restrict__ w16,
               float* __restrict__ out)
{
  const int lane = threadIdx.x & 31;
  const int half = lane >> 4, lm = lane & 15;
  const int NG = PIX / 64;                               // 128
  int wv = blockIdx.x * 8 + (threadIdx.x >> 5);          // 768 waves
  const int ng = wv % NG, mtile = wv / NG;               // mtile 0..5
  const int row0 = mtile << 4;
  const int p0 = ng << 6;

  v8f acc[4] = {};
  const _Float16* wrow = w16 + (size_t)(row0 + lm) * D_INNER;

  for (int kk = 0; kk < D_INNER; kk += 32) {
    v8h alo = *(const v8h*)(wrow + kk + half * 8);
    v8h ahi = *(const v8h*)(wrow + kk + 16 + half * 8);
    v16h a = joinA(alo, ahi);
#pragma unroll
    for (int t = 0; t < 4; ++t) {
      const _Float16* bp = g16 + (size_t)(p0 + t * 16 + lm) * D_INNER + kk + half * 16;
      __builtin_prefetch(bp + 32, 0, 0);
      v16h bb = joinA(*(const v8h*)(bp - half * 8), *(const v8h*)(bp - half * 8 + 16));
      acc[t] = __builtin_amdgcn_wmma_f32_16x16x32_f16(false, a, false, bb,
                                                      (short)0, acc[t], false, false);
    }
  }
#pragma unroll
  for (int t = 0; t < 4; ++t) {
    int p = p0 + t * 16 + lm;
    int b = p >> 12, l = p & (LL - 1);
#pragma unroll
    for (int j = 0; j < 8; ++j) {
      int o = row0 + j + half * 8;
      out[((size_t)b * D_MODEL + o) * LL + l] = acc[t][j];
    }
  }
}

// ---------------------------------------------------------------------------
// Host launch
// ---------------------------------------------------------------------------
extern "C" void kernel_launch(void* const* d_in, const int* in_sizes, int n_in,
                              void* d_out, int out_size, void* d_ws, size_t ws_size,
                              hipStream_t stream)
{
  const float* x      = (const float*)d_in[0];
  const float* lprior = (const float*)d_in[1];
  const float* omap   = (const float*)d_in[2];
  const float* umap   = (const float*)d_in[3];
  const float* inw    = (const float*)d_in[4];
  const float* cw     = (const float*)d_in[5];
  const float* cb     = (const float*)d_in[6];
  const float* xpw    = (const float*)d_in[7];
  const float* dtw    = (const float*)d_in[8];
  const float* dtb    = (const float*)d_in[9];
  const float* Alogs  = (const float*)d_in[10];
  const float* Dsp    = (const float*)d_in[11];
  const float* lnw    = (const float*)d_in[12];
  const float* lnb    = (const float*)d_in[13];
  const float* opw    = (const float*)d_in[14];
  const float* alpha  = (const float*)d_in[15];
  const float* lw     = (const float*)d_in[16];
  const float* lb     = (const float*)d_in[17];
  float* out = (float*)d_out;

  float* ws = (float*)d_ws;
  const size_t SZ = (size_t)NB * D_INNER * LL;          // 1,572,864 f32
  float* xx   = ws;  ws += SZ;                          // channel-major f32
  float* z1T  = ws;  ws += SZ;                          // pixel-major f32
  float* z2T  = ws;  ws += SZ;
  float* xcb  = ws;  ws += SZ;                          // channel-major f32
  float* yacT = ws;  ws += SZ;                          // pixel-major f32
  float* dtrT = ws;  ws += (size_t)NB * 4 * DT_RANK * LL;
  float* BsT  = ws;  ws += (size_t)NB * 4 * D_STATE * LL;
  float* CsT  = ws;  ws += (size_t)NB * 4 * D_STATE * LL;
  _Float16* xc16  = (_Float16*)ws;  ws += SZ / 2;       // pixel-major f16
  _Float16* g16   = (_Float16*)ws;  ws += SZ / 2;
  _Float16* xT    = (_Float16*)ws;  ws += (size_t)NB * D_MODEL * LL / 2;
  _Float16* inw16 = (_Float16*)ws;  ws += (3 * D_INNER * D_MODEL) / 2;
  _Float16* xpw16 = (_Float16*)ws;  ws += (4 * 48 * D_INNER) / 2;
  _Float16* opw16 = (_Float16*)ws;  ws += (D_MODEL * D_INNER) / 2;

  // --- staging / conversion (depend only on inputs) ---
  k_zero<<<(int)((SZ + 255) / 256), 256, 0, stream>>>(yacT, (int)SZ);
  k_f32tof16<<<(3 * D_INNER * D_MODEL + 255) / 256, 256, 0, stream>>>(
      inw, inw16, 3 * D_INNER * D_MODEL);
  k_f32tof16<<<(D_MODEL * D_INNER + 255) / 256, 256, 0, stream>>>(
      opw, opw16, D_MODEL * D_INNER);
  k_xpw_pad<<<(4 * 48 * D_INNER + 255) / 256, 256, 0, stream>>>(xpw, xpw16);
  k_xpose_in<<<(NB * D_MODEL * LL + 255) / 256, 256, 0, stream>>>(x, xT);

  // --- pipeline ---
  k_inproj<<<(36 * (PIX / 64)) / 8, 256, 0, stream>>>(xT, inw16, xx, z1T, z2T);
  k_conv<<<(int)((SZ + 255) / 256), 256, 0, stream>>>(xx, cw, cb, lprior, lw, lb,
                                                      xcb, xc16);
  k_xdbl<<<(NB * 4 * 3 * (LL / 64)) / 8, 256, 0, stream>>>(xc16, xpw16,
                                                           dtrT, BsT, CsT);
  k_scan<<<(NB * 4 * D_INNER / 2 * 32) / 256, 256, 0, stream>>>(
      xcb, dtrT, BsT, CsT, dtw, dtb, Alogs, Dsp, yacT);
  k_lngate<<<(PIX * 32) / 256, 256, 0, stream>>>(
      yacT, z1T, z2T, omap, umap, lw, lb, lnw, lnb, alpha, g16);
  k_outproj<<<(6 * (PIX / 64)) / 8, 256, 0, stream>>>(g16, opw16, out);
}
